// HadamardLayer_6390911336774
// MI455X (gfx1250) — matrix operand order repaired and checked by reading, not compiled, not run on hardware
//
#include <hip/hip_runtime.h>

// ---------------------------------------------------------------------------
// Fused Hadamard encode+decode:  yhat[n] = C * (C * y[n])   (C symmetric)
// y: [16, 256, 128, 128] f32,  C: [256, 256] f32,  out: same shape as y.
//
// Per workgroup (512 thr = 16 wave32s, 256KB LDS, 1 WG/WGP):
//   - stage C once as bf16 (128KB), then loop over 4 pixel-tile jobs:
//       stage Y tile transposed [pixel][ch] bf16 (64KB) + prefetch next tile
//       GEMM1: Z = C * Y   (bf16 WMMA, f32 acc, Z -> LDS bf16, 64KB)
//       GEMM2: out = C * Z (f32 stores to HBM)
// Each K-step issues all 16 ds_load_b128 B-fragment loads, then a scheduling
// fence (sched_barrier 0), then the 8 WMMAs -> deep dscnt pipelining instead
// of a full LDS-latency drain before every WMMA.
// ---------------------------------------------------------------------------

typedef __attribute__((ext_vector_type(16))) __bf16 v16bf;
typedef __attribute__((ext_vector_type(8)))  __bf16 v8bf;
typedef __attribute__((ext_vector_type(8)))  float  v8f;

#define CH 256            // channels (= Hadamard dimension)
#define HW 16384          // 128*128 pixels per (n, c) plane
#define PT 128            // pixel tile per job
#define JOBS 4            // pixel-tile jobs per workgroup
#define NTHREADS 512      // 16 wave32s
#define LDS_C_OFF 0
#define LDS_Y_OFF (CH * CH * 2)              // 131072
#define LDS_Z_OFF (LDS_Y_OFF + PT * CH * 2)  // 196608 ; total 262144 B

struct v8bf2 { v8bf lo, hi; };

static __device__ __forceinline__ void sched_fence() {
#if defined(__has_builtin)
#if __has_builtin(__builtin_amdgcn_sched_barrier)
  __builtin_amdgcn_sched_barrier(0);   // nothing may be reordered across
#else
  asm volatile("" ::: "memory");
#endif
#else
  asm volatile("" ::: "memory");
#endif
}

// A fragment (16x32 bf16, M x K) from a row-major [*][CH] bf16 buffer.
// Lanes 0-15 -> M = lane, K in {0..7, 16..23}; lanes 16-31 -> M = lane-16,
// K in {8..15, 24..31}.  Two contiguous 16B reads per lane.
static __device__ __forceinline__ v16bf loadA(const __bf16* base,
                                              int m0, int k0, int lane) {
  const int m  = m0 + (lane & 15);
  const int kb = k0 + ((lane >> 4) << 3);   // +8 for upper half-wave
  const __bf16* p = base + m * CH + kb;
  v8bf2 t;
  t.lo = *(const v8bf*)(p);        // K = kb .. kb+7
  t.hi = *(const v8bf*)(p + 16);   // K = kb+16 .. kb+23
  return __builtin_bit_cast(v16bf, t);
}

// B fragment (32x16 bf16, K x N) from buffer laid out [pixel][CH] (K-contig).
// Lanes 0-15 -> N = lane, K in {k0..k0+7, k0+16..k0+23} ; lanes 16-31 -> N = lane-16,
// K in {k0+8..k0+15, k0+24..k0+31}.  Two contiguous 16B reads per lane.
static __device__ __forceinline__ v16bf loadB(const __bf16* base,
                                              int n0, int k0, int lane) {
  const int p  = n0 + (lane & 15);
  const int kb = k0 + ((lane >> 4) << 3);   // +8 for upper half-wave
  const __bf16* q = base + p * CH + kb;
  v8bf2 t;
  t.lo = *(const v8bf*)(q);
  t.hi = *(const v8bf*)(q + 16);
  return __builtin_bit_cast(v16bf, t);
}

// Store a 16x16 f32 accumulator tile as bf16 into [pixel][CH] buffer.
// D layout: lane L holds N = L&15, M = r + 8*(L>>4), r = 0..7 -> the 8 values
// are K-consecutive -> one packed 16B LDS store per lane.
static __device__ __forceinline__ void storeZtile(__bf16* Zs, int m0, int n0,
                                                  int lane, v8f acc) {
  const int p  = n0 + (lane & 15);
  const int kb = m0 + ((lane >> 4) << 3);
  v8bf z;
#pragma unroll
  for (int r = 0; r < 8; ++r) z[r] = (__bf16)acc[r];
  *(v8bf*)(Zs + p * CH + kb) = z;
}

// Store a 16x16 f32 tile to global nchw output (16-lane / 64B coalesced rows).
static __device__ __forceinline__ void storeOut(float* __restrict__ out,
                                                size_t base, int m0, int n0,
                                                int lane, v8f acc) {
  const int p  = n0 + (lane & 15);
  const int cb = m0 + ((lane >> 4) << 3);
#pragma unroll
  for (int r = 0; r < 8; ++r)
    out[base + (size_t)(cb + r) * HW + p] = acc[r];
}

#define WMMA_BF16(a, b, c) \
  __builtin_amdgcn_wmma_f32_16x16x32_bf16(false, (a), false, (b), (short)0, (c), false, false)

// One 16x128 output stripe: 8 K-steps; per step, batch-issue all fragment
// loads, fence the scheduler, then fire the 8 WMMAs.
static __device__ __forceinline__ void gemm_stripe(const __bf16* Cs,
                                                   const __bf16* Bbuf,
                                                   int m0, int lane,
                                                   v8f acc[8]) {
#pragma unroll
  for (int nt = 0; nt < 8; ++nt) acc[nt] = (v8f){};
#pragma unroll
  for (int k = 0; k < CH; k += 32) {
    v16bf a = loadA(Cs, m0, k, lane);
    v16bf b0 = loadB(Bbuf,   0, k, lane);
    v16bf b1 = loadB(Bbuf,  16, k, lane);
    v16bf b2 = loadB(Bbuf,  32, k, lane);
    v16bf b3 = loadB(Bbuf,  48, k, lane);
    v16bf b4 = loadB(Bbuf,  64, k, lane);
    v16bf b5 = loadB(Bbuf,  80, k, lane);
    v16bf b6 = loadB(Bbuf,  96, k, lane);
    v16bf b7 = loadB(Bbuf, 112, k, lane);
    sched_fence();                       // keep all 18 ds_loads above the MMAs
    acc[0] = WMMA_BF16(a, b0, acc[0]);
    acc[1] = WMMA_BF16(a, b1, acc[1]);
    acc[2] = WMMA_BF16(a, b2, acc[2]);
    acc[3] = WMMA_BF16(a, b3, acc[3]);
    acc[4] = WMMA_BF16(a, b4, acc[4]);
    acc[5] = WMMA_BF16(a, b5, acc[5]);
    acc[6] = WMMA_BF16(a, b6, acc[6]);
    acc[7] = WMMA_BF16(a, b7, acc[7]);
  }
}

extern "C" __global__ __launch_bounds__(NTHREADS)
void hadamard_fused_wmma(const float* __restrict__ y,
                         const float* __restrict__ Cg,
                         float* __restrict__ out,
                         int jobs_total) {
  extern __shared__ char smem[];
  __bf16* Cs = (__bf16*)(smem + LDS_C_OFF);  // [256][256] bf16
  __bf16* Ys = (__bf16*)(smem + LDS_Y_OFF);  // [128][256] bf16 (pixel-major)
  __bf16* Zs = (__bf16*)(smem + LDS_Z_OFF);  // [128][256] bf16 (pixel-major)

  const int tid  = threadIdx.x;
  const int lane = tid & 31;
  const int wave = tid >> 5;                 // 0..15 -> owns m-rows wave*16..+15

  // ---- stage C once: fp32 -> bf16; thread t copies half a row (128 el) ----
  {
    const float* src = Cg + (tid >> 1) * CH + (tid & 1) * 128;
    __bf16*      dst = Cs + (tid >> 1) * CH + (tid & 1) * 128;
#pragma unroll 4
    for (int i = 0; i < 128; i += 8) {
      float4 a = *(const float4*)(src + i);
      float4 b = *(const float4*)(src + i + 4);
      v8bf pk;
      pk[0] = (__bf16)a.x; pk[1] = (__bf16)a.y;
      pk[2] = (__bf16)a.z; pk[3] = (__bf16)a.w;
      pk[4] = (__bf16)b.x; pk[5] = (__bf16)b.y;
      pk[6] = (__bf16)b.z; pk[7] = (__bf16)b.w;
      *(v8bf*)(dst + i) = pk;
    }
  }

  // ---- loop over pixel-tile jobs (amortizes C staging) --------------------
  for (int i = 0; i < JOBS; ++i) {
    const int job = blockIdx.x * JOBS + i;
    if (job >= jobs_total) break;            // uniform across block
    const int n  = job >> 7;                 // 128 tiles per batch
    const int p0 = (job & 127) * PT;

    // ---- stage Y tile: global [c][pixel] f32 -> LDS [pixel][c] bf16 -------
    {
      const int pg = tid & 3;                // 4 threads per channel row
#pragma unroll
      for (int pass = 0; pass < 2; ++pass) {
        const int c = pass * 128 + (tid >> 2);
        const float* src = y + ((size_t)n * CH + c) * HW + p0 + pg * 32;
#pragma unroll
        for (int j = 0; j < 32; j += 4) {
          float4 v = *(const float4*)(src + j);
          Ys[(pg * 32 + j + 0) * CH + c] = (__bf16)v.x;
          Ys[(pg * 32 + j + 1) * CH + c] = (__bf16)v.y;
          Ys[(pg * 32 + j + 2) * CH + c] = (__bf16)v.z;
          Ys[(pg * 32 + j + 3) * CH + c] = (__bf16)v.w;
        }
      }
      // prefetch next job's rows toward L2 while the two GEMMs run
      const int nxt = job + 1;
      if (nxt < jobs_total && (i + 1) < JOBS) {
        const int nn  = nxt >> 7;
        const int np0 = (nxt & 127) * PT;
#pragma unroll
        for (int pass = 0; pass < 2; ++pass) {
          const int c = pass * 128 + (tid >> 2);
          const float* nsrc = y + ((size_t)nn * CH + c) * HW + np0 + pg * 32;
          __builtin_prefetch(nsrc, 0, 1);
        }
      }
    }
    __syncthreads();

    // ---- GEMM1: Z[k, p] = sum_c C[c,k] * Y[c,p]  (C symmetric) ------------
    {
      const int m0 = wave * 16;
      v8f acc[8];
      gemm_stripe(Cs, Ys, m0, lane, acc);
#pragma unroll
      for (int nt = 0; nt < 8; ++nt) storeZtile(Zs, m0, nt * 16, lane, acc[nt]);
    }
    __syncthreads();

    // ---- GEMM2: yhat[c, p] = sum_k C[c,k] * Z[k,p] ; f32 out to HBM -------
    {
      const int m0 = wave * 16;
      const size_t obase = (size_t)n * CH * HW + p0;
      v8f acc[8];
      gemm_stripe(Cs, Zs, m0, lane, acc);
#pragma unroll
      for (int nt = 0; nt < 8; ++nt) storeOut(out, obase, m0, nt * 16, lane, acc[nt]);
    }
    __syncthreads();   // protect Ys/Zs before next job's staging
  }
}

extern "C" void kernel_launch(void* const* d_in, const int* in_sizes, int n_in,
                              void* d_out, int out_size, void* d_ws, size_t ws_size,
                              hipStream_t stream) {
  (void)n_in; (void)d_ws; (void)ws_size; (void)out_size;
  const float* y   = (const float*)d_in[0];
  const float* C   = (const float*)d_in[1];
  float*       out = (float*)d_out;

  const int nbatch     = in_sizes[0] / (CH * HW);     // 16
  const int jobs_total = nbatch * (HW / PT);          // 2048
  const int nblocks    = (jobs_total + JOBS - 1) / JOBS;  // 512
  const size_t shmem   = (size_t)(CH * CH * 2 + 2 * PT * CH * 2);  // 256 KB

  (void)hipFuncSetAttribute((const void*)hadamard_fused_wmma,
                            hipFuncAttributeMaxDynamicSharedMemorySize,
                            (int)shmem);

  hipLaunchKernelGGL(hadamard_fused_wmma,
                     dim3(nblocks), dim3(NTHREADS), shmem, stream,
                     y, C, out, jobs_total);
}
